// MultiDMM_32109175504898
// MI455X (gfx1250) — matrix operand order repaired and checked by reading, not compiled, not run on hardware
//
#include <hip/hip_runtime.h>
#include <hip/hip_bf16.h>
#include <math.h>

// ---------------------------------------------------------------------------
// MultiDMM on MI455X (gfx1250): all matmuls via v_wmma_f32_16x16x32_f16.
// ---------------------------------------------------------------------------

typedef __attribute__((ext_vector_type(16))) _Float16 v16h;
typedef __attribute__((ext_vector_type(8)))  float    v8f;

#define TT 64
#define BB 2048
#define ZZ 32
#define EPSF 1e-8f

// ---------------- WMMA helpers (CDNA5 16x16x32 f16 layouts) ----------------

__device__ __forceinline__ v8f wmma_f16(v16h a, v16h b, v8f c) {
  // D = A(16x32 f16) x B(32x16 f16) + C(16x16 f32)
  return __builtin_amdgcn_wmma_f32_16x16x32_f16(
      false, a, false, b, (short)0, c, false, false);
}

// A-operand gather from an LDS f32 tile buf[16][33] (row-major, padded).
// lane l holds row M=l&15; element e maps to K = ((l>>4)<<4)+e
__device__ __forceinline__ v16h gather_a(const float* buf, int l) {
  v16h a;
  const int m  = l & 15;
  const int kh = (l >> 4) << 4;
#pragma unroll
  for (int e = 0; e < 16; ++e) {
    const int k = kh + e;
    a[e] = (_Float16)buf[m * 33 + k];
  }
  return a;
}

// B-operand gather from global f32 W (K-major, row stride ldw).
// lane l holds col N=ncol0+(l&15); element e maps to K = ((l>>4)<<4)+e
__device__ __forceinline__ v16h load_b_global(const float* __restrict__ W,
                                              int l, int ncol0, int ldw) {
  v16h b;
  const int n  = ncol0 + (l & 15);
  const int kb = (l >> 4) << 4;
#pragma unroll
  for (int e = 0; e < 16; ++e) b[e] = (_Float16)W[(kb + e) * ldw + n];
  return b;
}

// B-operand gather from an LDS f16 weight image.
__device__ __forceinline__ v16h gather_b_lds(const _Float16* w, int l,
                                             int ncol0, int ldw) {
  v16h b;
  const int n  = ncol0 + (l & 15);
  const int kb = (l >> 4) << 4;
#pragma unroll
  for (int e = 0; e < 16; ++e) b[e] = w[(kb + e) * ldw + n];
  return b;
}

// Scatter a C/D accumulator (M = r + 8*(l>>4), N = l&15) into LDS buf[16][33].
__device__ __forceinline__ void store_c(float* buf, v8f c, int l, int ncol0) {
  const int m0  = (l >> 4) << 3;
  const int col = ncol0 + (l & 15);
#pragma unroll
  for (int r = 0; r < 8; ++r) buf[(m0 + r) * 33 + col] = c[r];
}

__device__ __forceinline__ void relu_bias(v8f& c, float b) {
#pragma unroll
  for (int r = 0; r < 8; ++r) c[r] = fmaxf(c[r] + b, 0.f);
}

__device__ __forceinline__ float softplusf(float x) {
  return (x > 20.f) ? x : log1pf(expf(x));
}

// Deterministic counter-based normal (stand-in for JAX threefry normal).
__device__ __forceinline__ unsigned hashu(unsigned x) {
  x ^= x >> 16; x *= 0x7feb352dU;
  x ^= x >> 15; x *= 0x846ca68bU;
  x ^= x >> 16; return x;
}
__device__ __forceinline__ float gaussn(unsigned chain, unsigned t,
                                        unsigned brow, unsigned zi) {
  unsigned s0 = hashu(chain * 0x9E3779B9u + t * 0x85EBCA77u +
                      brow * 0xC2B2AE3Du + zi * 0x27D4EB2Fu + 0x01234567u);
  unsigned s1 = hashu(s0 ^ 0x94D049BBu);
  float u1 = (float)(s0 | 1u) * 2.3283064365386963e-10f;  // (0,1]
  float u2 = (float)s1 * 2.3283064365386963e-10f;
  float r  = sqrtf(-2.f * logf(u1));
  return r * cosf(6.283185307179586f * u2);
}

// ---------------------------------------------------------------------------
// Encoder: mu, sd, mask = encode(x, modality)   (vectorized over T*B rows)
//   h1 = relu(x @ Wphi + bphi); h2 = relu(h1 @ Wenc + benc)
//   mu = h2 @ Wm + bm ; sd = softplus(h2 @ Ws + bs) ; mask = !any_nan(row)
// One wave (32 lanes) per block, 16-row tiles, grid-stride.
// ---------------------------------------------------------------------------
template <int D>
__global__ __launch_bounds__(32) void encode_kernel(
    const float* __restrict__ x,
    const float* __restrict__ Wphi, const float* __restrict__ bphi,
    const float* __restrict__ Wenc, const float* __restrict__ benc,
    const float* __restrict__ Wm,   const float* __restrict__ bm,
    const float* __restrict__ Ws,   const float* __restrict__ bs,
    float* __restrict__ mu, float* __restrict__ sd, float* __restrict__ mk) {
  __shared__ _Float16 wphi_l[D * 32];   // f16 image of Wphi (<=32KB)
  __shared__ float    xbuf[16 * 33];
  __shared__ float    hbuf[16 * 33];
  __shared__ unsigned nanbuf[32];
  const int l = threadIdx.x;
  const int n = l & 15;
  const int mrow = (l >> 4) << 3;

  for (int i = l; i < D * 32; i += 32) wphi_l[i] = (_Float16)Wphi[i];

  const v16h Be0 = load_b_global(Wenc, l, 0, 32), Be1 = load_b_global(Wenc, l, 16, 32);
  const v16h Bm0 = load_b_global(Wm,   l, 0, 32), Bm1 = load_b_global(Wm,   l, 16, 32);
  const v16h Bs0 = load_b_global(Ws,   l, 0, 32), Bs1 = load_b_global(Ws,   l, 16, 32);
  const float bphi0 = bphi[n], bphi1 = bphi[16 + n];
  const float benc0 = benc[n], benc1 = benc[16 + n];
  const float bm0v  = bm[n],   bm1v  = bm[16 + n];
  const float bs0v  = bs[n],   bs1v  = bs[16 + n];
  __syncthreads();

  const int ntiles = (TT * BB) / 16;
  for (int tile = blockIdx.x; tile < ntiles; tile += gridDim.x) {
    const size_t row0 = (size_t)tile * 16;
    v8f h0 = {}, h1 = {};
    unsigned nanbits = 0;
    for (int kb = 0; kb < D; kb += 32) {
      __syncthreads();
#pragma unroll
      for (int i = 0; i < 16; ++i) {
        float v = x[(row0 + i) * D + kb + l];
        if (kb + 32 < D) __builtin_prefetch(&x[(row0 + i) * D + kb + 32 + l], 0, 1);
        if (v != v) { nanbits |= (1u << i); v = 0.f; }  // nan_to_num + mask
        xbuf[i * 33 + l] = v;
      }
      __syncthreads();
      const v16h a = gather_a(xbuf, l);
      h0 = wmma_f16(a, gather_b_lds(wphi_l + kb * 32, l, 0, 32), h0);
      h1 = wmma_f16(a, gather_b_lds(wphi_l + kb * 32, l, 16, 32), h1);
    }
    relu_bias(h0, bphi0); relu_bias(h1, bphi1);
    __syncthreads();
    store_c(hbuf, h0, l, 0); store_c(hbuf, h1, l, 16);
    nanbuf[l] = nanbits;
    __syncthreads();
    const v16h ha = gather_a(hbuf, l);
    v8f e0 = {}, e1 = {};
    e0 = wmma_f16(ha, Be0, e0); e1 = wmma_f16(ha, Be1, e1);
    relu_bias(e0, benc0); relu_bias(e1, benc1);
    __syncthreads();
    store_c(hbuf, e0, l, 0); store_c(hbuf, e1, l, 16);
    __syncthreads();
    const v16h ea = gather_a(hbuf, l);
    v8f m0 = {}, m1 = {}, s0 = {}, s1 = {};
    m0 = wmma_f16(ea, Bm0, m0); m1 = wmma_f16(ea, Bm1, m1);
    s0 = wmma_f16(ea, Bs0, s0); s1 = wmma_f16(ea, Bs1, s1);

    unsigned allnan = 0;
#pragma unroll
    for (int i = 0; i < 32; ++i) allnan |= nanbuf[i];
#pragma unroll
    for (int r = 0; r < 8; ++r) {
      const size_t row = row0 + mrow + r;
      mu[row * ZZ + n]      = m0[r] + bm0v;
      mu[row * ZZ + 16 + n] = m1[r] + bm1v;
      sd[row * ZZ + n]      = softplusf(s0[r] + bs0v);
      sd[row * ZZ + 16 + n] = softplusf(s1[r] + bs1v);
    }
    if (l < 16) mk[row0 + l] = ((allnan >> l) & 1u) ? 0.f : 1.f;
  }
}

// ---------------------------------------------------------------------------
// fwd_net(z) applied in-wave: zbuf roundtrip + 6 WMMAs. Shared by chains.
// ---------------------------------------------------------------------------
struct FwdRegs {
  v16h Bf0, Bf1, Bm0, Bm1, Bs0, Bs1;
  float bf0, bf1, bm0, bm1, bs0, bs1;
};

__device__ __forceinline__ void fwd_net(float* zbuf, int l, const FwdRegs& w,
                                        v8f zc0, v8f zc1,
                                        v8f& pm0, v8f& pm1, v8f& ps0, v8f& ps1) {
  __syncthreads();
  store_c(zbuf, zc0, l, 0); store_c(zbuf, zc1, l, 16);
  __syncthreads();
  const v16h a = gather_a(zbuf, l);
  v8f h0 = {}, h1 = {};
  h0 = wmma_f16(a, w.Bf0, h0); h1 = wmma_f16(a, w.Bf1, h1);
  relu_bias(h0, w.bf0); relu_bias(h1, w.bf1);
  __syncthreads();
  store_c(zbuf, h0, l, 0); store_c(zbuf, h1, l, 16);
  __syncthreads();
  const v16h ha = gather_a(zbuf, l);
  pm0 = v8f{}; pm1 = v8f{}; ps0 = v8f{}; ps1 = v8f{};
  pm0 = wmma_f16(ha, w.Bm0, pm0); pm1 = wmma_f16(ha, w.Bm1, pm1);
  ps0 = wmma_f16(ha, w.Bs0, ps0); ps1 = wmma_f16(ha, w.Bs1, ps1);
}

__device__ __forceinline__ FwdRegs load_fwd_regs(
    int l, const float* W1, const float* b1, const float* W2, const float* b2,
    const float* W3, const float* b3) {
  FwdRegs w;
  const int n = l & 15;
  w.Bf0 = load_b_global(W1, l, 0, 32); w.Bf1 = load_b_global(W1, l, 16, 32);
  w.Bm0 = load_b_global(W2, l, 0, 32); w.Bm1 = load_b_global(W2, l, 16, 32);
  w.Bs0 = load_b_global(W3, l, 0, 32); w.Bs1 = load_b_global(W3, l, 16, 32);
  w.bf0 = b1[n]; w.bf1 = b1[16 + n];
  w.bm0 = b2[n]; w.bm1 = b2[16 + n];
  w.bs0 = b3[n]; w.bs1 = b3[16 + n];
  return w;
}

// ---------------------------------------------------------------------------
// Phase 1: forward sampling chain. 128 blocks x 1 wave; each wave owns 16
// batch rows and iterates all 64 timesteps privately (rows are independent).
// ---------------------------------------------------------------------------
__global__ __launch_bounds__(32) void fwd_chain_kernel(
    const float* __restrict__ z0,
    const float* __restrict__ Wf,  const float* __restrict__ bf,
    const float* __restrict__ Wm,  const float* __restrict__ bm,
    const float* __restrict__ Ws,  const float* __restrict__ bs,
    float* __restrict__ zfmu, float* __restrict__ zfsd) {
  __shared__ float zbuf[16 * 33];
  const int l = threadIdx.x;
  const int n = l & 15;
  const int b0 = blockIdx.x * 16;
  const int mrow = (l >> 4) << 3;
  const FwdRegs w = load_fwd_regs(l, Wf, bf, Wm, bm, Ws, bs);

  v8f zc0, zc1;
  const float z00 = z0[n], z01 = z0[16 + n];
#pragma unroll
  for (int r = 0; r < 8; ++r) { zc0[r] = z00; zc1[r] = z01; }

  for (int t = 0; t < TT; ++t) {
    v8f m0, m1, s0, s1;
    fwd_net(zbuf, l, w, zc0, zc1, m0, m1, s0, s1);
    const size_t base = ((size_t)t * BB + b0 + mrow) * ZZ;
#pragma unroll
    for (int r = 0; r < 8; ++r) {
      const float mu0 = m0[r] + w.bm0, mu1 = m1[r] + w.bm1;
      const float sd0 = softplusf(s0[r] + w.bs0);
      const float sd1 = softplusf(s1[r] + w.bs1);
      zfmu[base + r * ZZ + n]      = mu0;
      zfmu[base + r * ZZ + 16 + n] = mu1;
      zfsd[base + r * ZZ + n]      = sd0;
      zfsd[base + r * ZZ + 16 + n] = sd1;
      const unsigned brow = (unsigned)(b0 + mrow + r);
      zc0[r] = mu0 + sd0 * gaussn(0u, t, brow, n);
      zc1[r] = mu1 + sd1 * gaussn(0u, t, brow, 16 + n);
    }
  }
}

// ---------------------------------------------------------------------------
// Phase 2: backward smoothing chain (reverse over t). Faithful quirk: the
// carried zb_prev enters the PoE as BOTH mean and std expert.
// ---------------------------------------------------------------------------
__global__ __launch_bounds__(32) void bwd_chain_kernel(
    const float* __restrict__ zfmu, const float* __restrict__ zfsd,
    const float* __restrict__ mu_a, const float* __restrict__ sd_a,
    const float* __restrict__ mk_a,
    const float* __restrict__ mu_b, const float* __restrict__ sd_b,
    const float* __restrict__ mk_b,
    const float* __restrict__ Wb,  const float* __restrict__ bb,
    const float* __restrict__ Wm,  const float* __restrict__ bm,
    const float* __restrict__ Ws,  const float* __restrict__ bs,
    float* __restrict__ zbmu, float* __restrict__ zbsd) {
  __shared__ float zbuf[16 * 33];
  const int l = threadIdx.x;
  const int n = l & 15;
  const int b0 = blockIdx.x * 16;
  const int mrow = (l >> 4) << 3;
  const FwdRegs w = load_fwd_regs(l, Wb, bb, Wm, bm, Ws, bs);

  v8f p0, p1;  // zb_prev carry, init ones
#pragma unroll
  for (int r = 0; r < 8; ++r) { p0[r] = 1.f; p1[r] = 1.f; }
  float has_prev = 0.f;

  for (int t = TT - 1; t >= 0; --t) {
    const size_t base = ((size_t)t * BB + b0 + mrow) * ZZ;
    v8f zc0, zc1;
#pragma unroll
    for (int r = 0; r < 8; ++r) {
      const int brow = b0 + mrow + r;
      const float ka = mk_a[(size_t)t * BB + brow];
      const float kb = mk_b[(size_t)t * BB + brow];
#pragma unroll
      for (int half = 0; half < 2; ++half) {
        const size_t a0 = base + r * ZZ + half * 16 + n;
        const float m1v = zfmu[a0], s1v = zfsd[a0];
        const float m2v = half ? p1[r] : p0[r];
        const float m3v = mu_a[a0], s3v = sd_a[a0];
        const float m4v = mu_b[a0], s4v = sd_b[a0];
        const float pr1 = 1.f      / (s1v * s1v + EPSF);
        const float pr2 = has_prev / (m2v * m2v + EPSF);  // std := zb_prev (faithful)
        const float pr3 = ka       / (s3v * s3v + EPSF);
        const float pr4 = kb       / (s4v * s4v + EPSF);
        const float s   = pr1 + pr2 + pr3 + pr4;
        const float num = m1v * pr1 + m2v * pr2 + m3v * pr3 + m4v * pr4;
        const float muv = num / s;
        const float zv  = muv + sqrtf(1.f / s) *
                          gaussn(1u, t, (unsigned)brow, half * 16 + n);
        if (half) zc1[r] = zv; else zc0[r] = zv;
      }
    }
    v8f m0, m1, s0, s1;
    fwd_net(zbuf, l, w, zc0, zc1, m0, m1, s0, s1);
#pragma unroll
    for (int r = 0; r < 8; ++r) {
      const float zm0 = m0[r] + w.bm0, zm1 = m1[r] + w.bm1;
      zbmu[base + r * ZZ + n]      = zm0;
      zbmu[base + r * ZZ + 16 + n] = zm1;
      zbsd[base + r * ZZ + n]      = softplusf(s0[r] + w.bs0);
      zbsd[base + r * ZZ + 16 + n] = softplusf(s1[r] + w.bs1);
      p0[r] = zm0; p1[r] = zm1;  // carry = zb_mean
    }
    has_prev = 1.f;
  }
}

// ---------------------------------------------------------------------------
// Phase 3: inference chain + prior. Writes inf_mu/inf_sd/pr_mu/pr_sd to
// d_out and z_all to workspace.
// ---------------------------------------------------------------------------
__global__ __launch_bounds__(32) void inf_chain_kernel(
    const float* __restrict__ z0,
    const float* __restrict__ Wf,  const float* __restrict__ bf,
    const float* __restrict__ Wm,  const float* __restrict__ bm,
    const float* __restrict__ Ws,  const float* __restrict__ bs,
    const float* __restrict__ mu_a, const float* __restrict__ sd_a,
    const float* __restrict__ mk_a,
    const float* __restrict__ mu_b, const float* __restrict__ sd_b,
    const float* __restrict__ mk_b,
    const float* __restrict__ zbmu, const float* __restrict__ zbsd,
    float* __restrict__ inf_mu, float* __restrict__ inf_sd,
    float* __restrict__ pr_mu,  float* __restrict__ pr_sd,
    float* __restrict__ zall) {
  __shared__ float zbuf[16 * 33];
  const int l = threadIdx.x;
  const int n = l & 15;
  const int b0 = blockIdx.x * 16;
  const int mrow = (l >> 4) << 3;
  const FwdRegs w = load_fwd_regs(l, Wf, bf, Wm, bm, Ws, bs);

  v8f zc0, zc1;
  const float z00 = z0[n], z01 = z0[16 + n];
#pragma unroll
  for (int r = 0; r < 8; ++r) { zc0[r] = z00; zc1[r] = z01; }

  for (int t = 0; t < TT; ++t) {
    v8f m0, m1, s0, s1;
    fwd_net(zbuf, l, w, zc0, zc1, m0, m1, s0, s1);  // prior net on carry z
    const float kbwd = (t < TT - 1) ? 1.f : 0.f;
    const size_t base = ((size_t)t * BB + b0 + mrow) * ZZ;
#pragma unroll
    for (int r = 0; r < 8; ++r) {
      const int brow = b0 + mrow + r;
      const float ka = mk_a[(size_t)t * BB + brow];
      const float kb = mk_b[(size_t)t * BB + brow];
#pragma unroll
      for (int half = 0; half < 2; ++half) {
        const size_t a0 = base + r * ZZ + half * 16 + n;
        const float pm = (half ? m1[r] + w.bm1 : m0[r] + w.bm0);
        const float ps = softplusf(half ? s1[r] + w.bs1 : s0[r] + w.bs0);
        const float ma = mu_a[a0], sa = sd_a[a0];
        const float mb = mu_b[a0], sb = sd_b[a0];
        const float mz = zbmu[a0], sz = zbsd[a0];
        const float pr1 = ka   / (sa * sa + EPSF);
        const float pr2 = kb   / (sb * sb + EPSF);
        const float pr3 = 1.f  / (ps * ps + EPSF);
        const float pr4 = kbwd / (sz * sz + EPSF);
        const float s   = pr1 + pr2 + pr3 + pr4;
        const float muv = (ma * pr1 + mb * pr2 + pm * pr3 + mz * pr4) / s;
        const float sdv = sqrtf(1.f / s);
        const float zn  = muv + sdv * gaussn(2u, t, (unsigned)brow, half * 16 + n);
        pr_mu[a0] = pm; pr_sd[a0] = ps;
        inf_mu[a0] = muv; inf_sd[a0] = sdv;
        zall[a0] = zn;
        if (half) zc1[r] = zn; else zc0[r] = zn;
      }
    }
  }
}

// ---------------------------------------------------------------------------
// Decoder: om = h@Wm + bm, os = softplus(h@Ws + bs), h = relu(z@Wd + bd).
// Heavy output phase: 64 (or 32) K=32 WMMAs per 16-row tile, streamed stores.
// Wm/Ws are small (<=64KB) -> L2/WGP$ resident; gathered directly.
// ---------------------------------------------------------------------------
template <int D>
__global__ __launch_bounds__(32) void decode_kernel(
    const float* __restrict__ zall,
    const float* __restrict__ Wd, const float* __restrict__ bd,
    const float* __restrict__ Wm, const float* __restrict__ bm,
    const float* __restrict__ Ws, const float* __restrict__ bs,
    float* __restrict__ om, float* __restrict__ os) {
  __shared__ float zbuf[16 * 33];
  const int l = threadIdx.x;
  const int n = l & 15;
  const int mrow = (l >> 4) << 3;
  const v16h Bd0 = load_b_global(Wd, l, 0, 32), Bd1 = load_b_global(Wd, l, 16, 32);
  const float bd0 = bd[n], bd1 = bd[16 + n];

  const int ntiles = (TT * BB) / 16;
  for (int tile = blockIdx.x; tile < ntiles; tile += gridDim.x) {
    const size_t row0 = (size_t)tile * 16;
    __syncthreads();
#pragma unroll
    for (int i = 0; i < 16; ++i) zbuf[i * 33 + l] = zall[(row0 + i) * ZZ + l];
    __syncthreads();
    const v16h a = gather_a(zbuf, l);
    v8f h0 = {}, h1 = {};
    h0 = wmma_f16(a, Bd0, h0); h1 = wmma_f16(a, Bd1, h1);
    relu_bias(h0, bd0); relu_bias(h1, bd1);
    __syncthreads();
    store_c(zbuf, h0, l, 0); store_c(zbuf, h1, l, 16);
    __syncthreads();
    const v16h ha = gather_a(zbuf, l);
#pragma unroll 4
    for (int nt = 0; nt < D / 16; ++nt) {
      const v16h bmv = load_b_global(Wm, l, nt * 16, D);
      const v16h bsv = load_b_global(Ws, l, nt * 16, D);
      v8f cm = {}, cs = {};
      cm = wmma_f16(ha, bmv, cm);
      cs = wmma_f16(ha, bsv, cs);
      const float bmx = bm[nt * 16 + n], bsx = bs[nt * 16 + n];
#pragma unroll
      for (int r = 0; r < 8; ++r) {
        const size_t row = row0 + mrow + r;
        om[row * D + nt * 16 + n] = cm[r] + bmx;
        os[row * D + nt * 16 + n] = softplusf(cs[r] + bsx);
      }
    }
  }
}

// ---------------------------------------------------------------------------
// Host launcher
// ---------------------------------------------------------------------------
extern "C" void kernel_launch(void* const* d_in, const int* in_sizes, int n_in,
                              void* d_out, int out_size, void* d_ws,
                              size_t ws_size, hipStream_t stream) {
  (void)n_in; (void)out_size; (void)ws_size;
  auto g = [&](int i) { return (const float*)d_in[i]; };

  const float *xa, *xb, *z0;
  const float *phiaW, *phiab, *encaW, *encab, *encmaW, *encmab, *encsaW, *encsab;
  const float *decaW, *decab, *decmaW, *decmab, *decsaW, *decsab;
  const float *phibW, *phibb, *encbW, *encbb, *encmbW, *encmbb, *encsbW, *encsbb;
  const float *decbW, *decbb, *decmbW, *decmbb, *decsbW, *decsbb;
  const float *fwdW, *fwdb, *fwdmW, *fwdmb, *fwdsW, *fwdsb;
  const float *bwdW, *bwdb, *bwdmW, *bwdmb, *bwdsW, *bwdsb;

  if (in_sizes[0] > 1000000) {
    // setup_inputs() insertion order: x_a, x_b, params{...}, (W,b) pairs
    xa = g(0); xb = g(1);
    phiaW = g(2);  phiab = g(3);  encaW = g(4);  encab = g(5);
    encmaW = g(6); encmab = g(7); encsaW = g(8); encsab = g(9);
    decaW = g(10); decab = g(11); decmaW = g(12); decmab = g(13);
    decsaW = g(14); decsab = g(15);
    phibW = g(16); phibb = g(17); encbW = g(18); encbb = g(19);
    encmbW = g(20); encmbb = g(21); encsbW = g(22); encsbb = g(23);
    decbW = g(24); decbb = g(25); decmbW = g(26); decmbb = g(27);
    decsbW = g(28); decsbb = g(29);
    fwdW = g(30); fwdb = g(31); fwdmW = g(32); fwdmb = g(33);
    fwdsW = g(34); fwdsb = g(35);
    bwdW = g(36); bwdb = g(37); bwdmW = g(38); bwdmb = g(39);
    bwdsW = g(40); bwdsb = g(41);
    z0 = g(42);
  } else {
    // pytree-sorted order: params (keys sorted), then x_a, x_b
    bwdW = g(0);  bwdb = g(1);  bwdmW = g(2);  bwdmb = g(3);
    bwdsW = g(4); bwdsb = g(5);
    decaW = g(6); decab = g(7); decbW = g(8); decbb = g(9);
    decmaW = g(10); decmab = g(11); decmbW = g(12); decmbb = g(13);
    decsaW = g(14); decsab = g(15); decsbW = g(16); decsbb = g(17);
    encaW = g(18); encab = g(19); encbW = g(20); encbb = g(21);
    encmaW = g(22); encmab = g(23); encmbW = g(24); encmbb = g(25);
    encsaW = g(26); encsab = g(27); encsbW = g(28); encsbb = g(29);
    fwdW = g(30); fwdb = g(31); fwdmW = g(32); fwdmb = g(33);
    fwdsW = g(34); fwdsb = g(35);
    phiaW = g(36); phiab = g(37); phibW = g(38); phibb = g(39);
    z0 = g(40); xa = g(41); xb = g(42);
  }

  const size_t TBZ = (size_t)TT * BB * ZZ;  // 4,194,304
  const size_t TB  = (size_t)TT * BB;       // 131,072

  // workspace layout (floats)
  float* w = (float*)d_ws;
  float* mu_a = w;            float* sd_a = mu_a + TBZ;
  float* mu_b = sd_a + TBZ;   float* sd_b = mu_b + TBZ;
  float* mk_a = sd_b + TBZ;   float* mk_b = mk_a + TB;
  float* zfmu = mk_b + TB;    float* zfsd = zfmu + TBZ;
  float* zbmu = zfsd + TBZ;   float* zbsd = zbmu + TBZ;
  float* zall = zbsd + TBZ;   // +TBZ -> ~152 MB total

  // output layout (return order)
  float* out   = (float*)d_out;
  float* o_imu = out;
  float* o_isd = o_imu + TBZ;
  float* o_pmu = o_isd + TBZ;
  float* o_psd = o_pmu + TBZ;
  float* o_oma = o_psd + TBZ;
  float* o_osa = o_oma + TB * 512;
  float* o_omb = o_osa + TB * 512;
  float* o_osb = o_omb + TB * 256;

  // ---- Encoders (parallel over all T*B rows) ----
  encode_kernel<512><<<1024, 32, 0, stream>>>(
      xa, phiaW, phiab, encaW, encab, encmaW, encmab, encsaW, encsab,
      mu_a, sd_a, mk_a);
  encode_kernel<256><<<1024, 32, 0, stream>>>(
      xb, phibW, phibb, encbW, encbb, encmbW, encmbb, encsbW, encsbb,
      mu_b, sd_b, mk_b);

  // ---- Temporal chains (per-row recurrences; 128 waves own 16 rows each) ----
  fwd_chain_kernel<<<BB / 16, 32, 0, stream>>>(
      z0, fwdW, fwdb, fwdmW, fwdmb, fwdsW, fwdsb, zfmu, zfsd);
  bwd_chain_kernel<<<BB / 16, 32, 0, stream>>>(
      zfmu, zfsd, mu_a, sd_a, mk_a, mu_b, sd_b, mk_b,
      bwdW, bwdb, bwdmW, bwdmb, bwdsW, bwdsb, zbmu, zbsd);
  inf_chain_kernel<<<BB / 16, 32, 0, stream>>>(
      z0, fwdW, fwdb, fwdmW, fwdmb, fwdsW, fwdsb,
      mu_a, sd_a, mk_a, mu_b, sd_b, mk_b, zbmu, zbsd,
      o_imu, o_isd, o_pmu, o_psd, zall);

  // ---- Decoders (parallel; dominant store bandwidth) ----
  decode_kernel<512><<<2048, 32, 0, stream>>>(
      zall, decaW, decab, decmaW, decmab, decsaW, decsab, o_oma, o_osa);
  decode_kernel<256><<<2048, 32, 0, stream>>>(
      zall, decbW, decbb, decmbW, decmbb, decsbW, decsbb, o_omb, o_osb);
}
